// SPSSConv3dTestTorch_71313636983477
// MI455X (gfx1250) — matrix operand order repaired and checked by reading, not compiled, not run on hardware
//
#include <hip/hip_runtime.h>
#include <hip/hip_bf16.h>

typedef __attribute__((ext_vector_type(16))) _Float16 v16h;
typedef __attribute__((ext_vector_type(8)))  _Float16 v8h;
typedef __attribute__((ext_vector_type(8)))  float    v8f;

#define NV    262144
#define CIN   64
#define COUT  64
#define DD    41
#define HH    400
#define WW    400
#define HCAP  (1 << 20)          // hash capacity (25% load factor)
#define HMASK (HCAP - 1)
#define WLAY_ELEMS (27 * 2 * 4 * 32 * 16)   // k, kchunk, outchunk, lane, 16 halves

__device__ __forceinline__ unsigned int hash_code(int code) {
    return (((unsigned int)code) * 2654435761u) >> 12;  // top 20 bits
}

// ---------------- Phase 1: init hash keys ----------------
__global__ void init_hash_kernel(int* __restrict__ hkeys) {
    int i = blockIdx.x * blockDim.x + threadIdx.x;
    if (i < HCAP) hkeys[i] = -1;
}

// ---------------- Phase 2: insert voxels ----------------
__global__ void insert_kernel(const int* __restrict__ coors,
                              int* __restrict__ hkeys,
                              int* __restrict__ hvals) {
    int i = blockIdx.x * blockDim.x + threadIdx.x;
    if (i >= NV) return;
    int b = coors[i * 4 + 0];
    int z = coors[i * 4 + 1];
    int y = coors[i * 4 + 2];
    int x = coors[i * 4 + 3];
    int code = ((b * DD + z) * HH + y) * WW + x;
    unsigned int slot = hash_code(code) & HMASK;
    while (true) {
        int prev = atomicCAS(&hkeys[slot], -1, code);
        if (prev == -1 || prev == code) { hvals[slot] = i; break; }
        slot = (slot + 1) & HMASK;
    }
}

// ---------------- Phase 3: repack weights to f16 B-fragment layout ----------
// wlay[(((k*2 + kc)*4 + oc)*32 + lane)*16 + i] = W[k][kc*32 + 8*(lane>>4) + (i&7) + 16*(i>>3)][oc*16 + (lane&15)]
__global__ void pack_weight_kernel(const float* __restrict__ weight,
                                   _Float16* __restrict__ wlay) {
    int idx = blockIdx.x * blockDim.x + threadIdx.x;
    if (idx >= WLAY_ELEMS) return;
    int i    = idx & 15;
    int lane = (idx >> 4) & 31;
    int oc   = (idx >> 9) & 3;
    int kc   = (idx >> 11) & 1;
    int k    = idx >> 12;          // 0..26
    int K    = kc * 32 + 8 * (lane >> 4) + (i & 7) + ((i >> 3) << 4);
    int col  = oc * 16 + (lane & 15);
    wlay[idx] = (_Float16)weight[k * (CIN * COUT) + K * COUT + col];
}

// ---------------- Phase 4: gather + WMMA conv (double-buffered) ------------
// One block = 16 output voxels. 128 threads = 4 waves; wave w owns out cols [16w,16w+16).
__global__ void __launch_bounds__(128)
spss_conv_wmma_kernel(const float* __restrict__ feats,
                      const int* __restrict__ coors,
                      const unsigned char* __restrict__ mask,
                      const int* __restrict__ hkeys,
                      const int* __restrict__ hvals,
                      const _Float16* __restrict__ wlay,
                      float* __restrict__ out) {
    __shared__ int       nbr[27 * 16];
    __shared__ _Float16  atile[2][16][72];   // double buffer; 144B row stride

    const int tid  = threadIdx.x;
    const int lane = tid & 31;
    const int wave = tid >> 5;
    const int base = blockIdx.x * 16;

    // ---- rulebook: neighbor row index per (offset k, tile row t) ----
    for (int idx = tid; idx < 27 * 16; idx += 128) {
        int k = idx >> 4;
        int t = idx & 15;
        int v = base + t;
        int res = -1;
        if (k == 13) {
            res = v;                               // center: always self
        } else if (mask[v]) {                      // pruned voxels: center-only
            int dz = k / 9 - 1;
            int dy = (k / 3) % 3 - 1;
            int dx = k % 3 - 1;
            int b = coors[v * 4 + 0];
            int z = coors[v * 4 + 1] + dz;
            int y = coors[v * 4 + 2] + dy;
            int x = coors[v * 4 + 3] + dx;
            if (z >= 0 && z < DD && y >= 0 && y < HH && x >= 0 && x < WW) {
                int code = ((b * DD + z) * HH + y) * WW + x;
                unsigned int slot = hash_code(code) & HMASK;
                while (true) {
                    int kk = hkeys[slot];
                    if (kk == code) { res = hvals[slot]; break; }
                    if (kk == -1) break;
                    slot = (slot + 1) & HMASK;
                }
            }
        }
        nbr[idx] = res;
    }

    // per-thread invariants
    const int srow = tid >> 3;                     // staging row 0..15
    const int scg  = (tid & 7) * 8;                // staging col group (8 ch)
    const int arow     = lane & 15;                // A frag: row
    const int acolbase = (lane >> 4) * 8;          // A frag: K base per half-wave

    v8f acc = {};                                  // 16x16 f32 accumulator

    // staging helper (lambda keeps EXEC divergence away from WMMAs)
    auto stage = [&](int k, int buf) {
        int src = nbr[k * 16 + srow];
        float4 f0 = make_float4(0.f, 0.f, 0.f, 0.f);
        float4 f1 = f0;
        if (src >= 0) {
            const float4* p = (const float4*)(feats + (long)src * CIN + scg);
            f0 = p[0];
            f1 = p[1];
        }
        _Float16* d = &atile[buf][srow][scg];
        d[0] = (_Float16)f0.x; d[1] = (_Float16)f0.y;
        d[2] = (_Float16)f0.z; d[3] = (_Float16)f0.w;
        d[4] = (_Float16)f1.x; d[5] = (_Float16)f1.y;
        d[6] = (_Float16)f1.z; d[7] = (_Float16)f1.w;
    };

    __syncthreads();           // rulebook visible
    stage(0, 0);               // prime the pipeline

    for (int k = 0; k < 27; ++k) {
        __syncthreads();       // buf[k&1] staged; compute on buf[(k-1)&1] finished
        if (k + 1 < 27) stage(k + 1, (k + 1) & 1);   // overlap next gather

        const int buf = k & 1;
        #pragma unroll
        for (int kc = 0; kc < 2; ++kc) {
            // A fragment per 16-bit A layout: K = 8*(lane>>4) + {0..7, 16..23}
            v8h lo = *(const v8h*)&atile[buf][arow][kc * 32 + acolbase];
            v8h hi = *(const v8h*)&atile[buf][arow][kc * 32 + acolbase + 16];
            v16h a = __builtin_shufflevector(lo, hi, 0, 1, 2, 3, 4, 5, 6, 7,
                                                      8, 9, 10, 11, 12, 13, 14, 15);
            // B fragment: contiguous 32B per lane from repacked weights
            const _Float16* wp = wlay + ((((k * 2 + kc) * 4 + wave) * 32 + lane) * 16);
            v16h b = *(const v16h*)wp;
            acc = __builtin_amdgcn_wmma_f32_16x16x32_f16(
                false, a, false, b, (short)0, acc, false, false);
        }
    }

    // ---- store C/D: VGPR r -> M = r + 8*(lane>>4), N = lane&15 (+ wave chunk) ----
    const int mbase = (lane >> 4) * 8;
    const int col   = wave * 16 + (lane & 15);
    #pragma unroll
    for (int r = 0; r < 8; ++r) {
        out[(long)(base + mbase + r) * COUT + col] = acc[r];
    }
}

// ---------------- host launcher ----------------
extern "C" void kernel_launch(void* const* d_in, const int* in_sizes, int n_in,
                              void* d_out, int out_size, void* d_ws, size_t ws_size,
                              hipStream_t stream) {
    const float*         feats  = (const float*)d_in[0];          // (N, 64) f32
    const int*           coors  = (const int*)d_in[1];            // (N, 4) i32
    const unsigned char* mask   = (const unsigned char*)d_in[2];  // (N,) bool (1B)
    const float*         weight = (const float*)d_in[3];          // (27, 64, 64) f32
    float*               out    = (float*)d_out;                  // (N, 64) f32

    // workspace layout: hash keys | hash vals | f16 packed weights  (~8.25 MB)
    int*      hkeys = (int*)d_ws;
    int*      hvals = hkeys + HCAP;
    _Float16* wlay  = (_Float16*)(hvals + HCAP);

    init_hash_kernel<<<HCAP / 256, 256, 0, stream>>>(hkeys);
    insert_kernel<<<(NV + 255) / 256, 256, 0, stream>>>(coors, hkeys, hvals);
    pack_weight_kernel<<<(WLAY_ELEMS + 255) / 256, 256, 0, stream>>>(weight, wlay);
    spss_conv_wmma_kernel<<<NV / 16, 128, 0, stream>>>(feats, coors, mask,
                                                       hkeys, hvals, wlay, out);
}
